// OuterProductMean_1279900254330
// MI455X (gfx1250) — matrix operand order repaired and checked by reading, not compiled, not run on hardware
//
#include <hip/hip_runtime.h>
#include <hip/hip_bf16.h>

typedef __attribute__((ext_vector_type(16))) __bf16        v16bf;
typedef __attribute__((ext_vector_type(8)))  float         v8f;
typedef __attribute__((ext_vector_type(8)))  unsigned int  v8u;
typedef __attribute__((ext_vector_type(2)))  float         f32x2;
typedef int v4i_ __attribute__((__vector_size__(16)));     // matches async-LDS builtin param type

#define S_DIM 128
#define I_DIM 256
#define CM    256
#define C_DIM 32
#define CZ    128
#define BI    16
#define BJ    16
#define LSTR  136   // padded LDS row stride (bf16 elements): rows stay 16B-aligned, banks spread

// hardware RNE f32->bf16 via native __bf16 convert
__device__ __forceinline__ unsigned short f2bfh(float f) {
    return __builtin_bit_cast(unsigned short, (__bf16)f);
}

// ---- async global->LDS 16B segment copy (CDNA5 path), with portable fallback -------
__device__ __forceinline__ void copy16_g2l(unsigned short* l, const unsigned short* g) {
#if __has_builtin(__builtin_amdgcn_global_load_async_to_lds_b128)
    __builtin_amdgcn_global_load_async_to_lds_b128(
        (__attribute__((address_space(1))) v4i_*)(unsigned short*)g,
        (__attribute__((address_space(3))) v4i_*)l, 0, 0);
#else
    *(uint4*)l = *(const uint4*)g;
#endif
}
// wait until at most N async ops remain outstanding (in-order completion => earlier ones done)
__device__ __forceinline__ void wait_g2l_le6() {
#if __has_builtin(__builtin_amdgcn_global_load_async_to_lds_b128)
#if __has_builtin(__builtin_amdgcn_s_wait_asynccnt)
    __builtin_amdgcn_s_wait_asynccnt(6);
#else
    asm volatile("s_wait_asynccnt 6" ::: "memory");
#endif
#endif
}
__device__ __forceinline__ void wait_g2l_le0() {
#if __has_builtin(__builtin_amdgcn_global_load_async_to_lds_b128)
#if __has_builtin(__builtin_amdgcn_s_wait_asynccnt)
    __builtin_amdgcn_s_wait_asynccnt(0);
#else
    asm volatile("s_wait_asynccnt 0" ::: "memory");
#endif
#endif
}

// ---------------- Kernel 0: interleave w_left/w_right -> Wlr[k*32+c] = {wl, wr} --------
__global__ __launch_bounds__(256) void wlr_pack_kernel(
    const float* __restrict__ wl, const float* __restrict__ wr, float* __restrict__ Wlr)
{
    int t = blockIdx.x*256 + threadIdx.x;       // 8192 total
    Wlr[2*t+0] = wl[t];
    Wlr[2*t+1] = wr[t];
}

// ---------------- Kernel 1: LayerNorm + dual projection --------------------------------
// Writes A,B TRANSPOSED: At[(i*32+c)][s], Bt[(j*32+d)][s] (bf16, s-contiguous rows) so
// kernel 2 can stage with contiguous async b128 copies. A pre-scaled by 1/S (2^-7 exact).
__global__ __launch_bounds__(256) void ln_proj_kernel(
    const float* __restrict__ msa, const float* __restrict__ ln_w, const float* __restrict__ ln_b,
    const float* __restrict__ Wlr,   // interleaved {w_left, w_right} per (k,c)
    const float* __restrict__ b_left, const float* __restrict__ b_right,
    unsigned short* __restrict__ At, unsigned short* __restrict__ Bt)
{
    const int wv   = threadIdx.x >> 5;
    const int lane = threadIdx.x & 31;
    const int i    = blockIdx.x;          // 0..255
    const int sg   = blockIdx.y;          // 0..15 (s-group of 8)
    const int s    = sg*8 + wv;
    const float* xr = msa + ((size_t)s*I_DIM + i)*CM;

    float x[8];
    float sum = 0.f;
#pragma unroll
    for (int k = 0; k < 8; ++k) { x[k] = xr[lane + 32*k]; sum += x[k]; }
#pragma unroll
    for (int m = 16; m >= 1; m >>= 1) sum += __shfl_xor(sum, m, 32);
    const float mu = sum * (1.0f/CM);
    float sq = 0.f;
#pragma unroll
    for (int k = 0; k < 8; ++k) { float d = x[k]-mu; sq += d*d; }
#pragma unroll
    for (int m = 16; m >= 1; m >>= 1) sq += __shfl_xor(sq, m, 32);
    const float rstd = rsqrtf(sq * (1.0f/CM) + 1e-5f);

    __shared__ float xn[8][CM];
    __shared__ float Asm[8][C_DIM];
    __shared__ float Bsm[8][C_DIM];
#pragma unroll
    for (int k = 0; k < 8; ++k) {
        int kk = lane + 32*k;
        xn[wv][kk] = (x[k]-mu)*rstd*ln_w[kk] + ln_b[kk];
    }
    __syncthreads();

    f32x2 ab;
    ab.x = b_left[lane];
    ab.y = b_right[lane];
    const f32x2* wp = ((const f32x2*)Wlr) + lane;   // wp[k*32] = {wl[k,lane], wr[k,lane]}
#pragma unroll 4
    for (int k = 0; k < CM; ++k) {
        float xv = xn[wv][k];
        f32x2 xv2 = {xv, xv};
        ab = __builtin_elementwise_fma(xv2, wp[k*C_DIM], ab);   // -> v_pk_fma_f32
    }
    Asm[wv][lane] = ab.x * (1.0f/S_DIM);
    Bsm[wv][lane] = ab.y;
    __syncthreads();

    // transpose 8(s) x 32(c) -> 16B-packed stores into At/Bt rows
    if (threadIdx.x < 64) {
        const int half = threadIdx.x >> 5;      // 0 = A, 1 = B
        const int c    = threadIdx.x & 31;
        float (*src)[C_DIM] = half ? Bsm : Asm;
        unsigned short* dst = half ? Bt : At;
        union { unsigned short u16[8]; uint4 q; } pk;
#pragma unroll
        for (int ss = 0; ss < 8; ++ss) pk.u16[ss] = f2bfh(src[ss][c]);
        *(uint4*)(dst + ((size_t)i*C_DIM + c)*S_DIM + sg*8) = pk.q;
    }
}

// ---------------- Kernel 1b: transpose w_out (f32 [k2][z]) -> Wt (bf16 [z][k2]) --------
__global__ __launch_bounds__(256) void wout_tr_kernel(
    const float* __restrict__ w_out, unsigned short* __restrict__ Wt)
{
    __shared__ float tile[32][33];
    const int tx = threadIdx.x & 31, ty = threadIdx.x >> 5;   // 32 x 8
    const int K0 = blockIdx.x * 32;   // k2 tile (of 1024)
    const int Z0 = blockIdx.y * 32;   // z  tile (of 128)
#pragma unroll
    for (int q = 0; q < 4; ++q) {
        int row = ty + 8*q;
        tile[row][tx] = w_out[(size_t)(K0+row)*CZ + Z0 + tx];
    }
    __syncthreads();
#pragma unroll
    for (int q = 0; q < 4; ++q) {
        int row = ty + 8*q;
        Wt[(size_t)(Z0+row)*(C_DIM*C_DIM) + K0 + tx] = f2bfh(tile[tx][row]);
    }
}

// ---- WMMA operand builders (bf16, per CDNA5 ISA layouts) -------------------------
// A 16x32: lane(h,r): M=r; VGPR v<4 -> K=2v+8h ; v>=4 -> K=16+2(v-4)+8h  (two 16B runs)
__device__ __forceinline__ v16bf load_a_op(const unsigned short* block, int r, int h, int k0) {
    const unsigned short* p = block + r*LSTR;
    uint4 q0 = *(const uint4*)(p + k0 + 8*h);
    uint4 q1 = *(const uint4*)(p + k0 + 16 + 8*h);
    v8u raw;
    raw[0]=q0.x; raw[1]=q0.y; raw[2]=q0.z; raw[3]=q0.w;
    raw[4]=q1.x; raw[5]=q1.y; raw[6]=q1.z; raw[7]=q1.w;
    return __builtin_bit_cast(v16bf, raw);
}
// B 32x16: lane(h,r): N=r; VGPR v<4 -> K=2v+8h ; v>=4 -> K=16+2(v-4)+8h  (two 16B runs)
__device__ __forceinline__ v16bf load_b_op(const unsigned short* block, int r, int h, int k0) {
    const unsigned short* p = block + r*LSTR + k0 + 8*h;
    uint4 q0 = *(const uint4*)(p);
    uint4 q1 = *(const uint4*)(p + 16);
    v8u raw;
    raw[0]=q0.x; raw[1]=q0.y; raw[2]=q0.z; raw[3]=q0.w;
    raw[4]=q1.x; raw[5]=q1.y; raw[6]=q1.z; raw[7]=q1.w;
    return __builtin_bit_cast(v16bf, raw);
}

// ---------------- Kernel 2: fused outer-product-mean + output projection ---------------
// Workgroup = 16 waves, owns a BIxBJ (i,j) block. Per c-chunk (4 c's):
//   Phase A: P[(i,j),(c,d)] = sum_s A[s,i,c]*B[s,j,d]  (WMMA, K=s=128) -> LDS (bf16)
//   Phase B: out[(i,j),z]  += P @ w_out_chunk          (WMMA, K=128)  -> f32 regs
// A-chunk and W-chunk LDS buffers are DOUBLE-BUFFERED: chunk cc+1 is fetched with async
// b128 copies while chunk cc computes; s_wait_asynccnt 6 drains only the older copies.
__global__ __launch_bounds__(512) void fused_opm_kernel(
    const unsigned short* __restrict__ At,   // [(i*32+c)][s] bf16, pre-scaled 1/s
    const unsigned short* __restrict__ Bt,   // [(j*32+d)][s] bf16
    const unsigned short* __restrict__ Wt,   // [z][k2=1024]  bf16
    const float* __restrict__ b_out,
    float* __restrict__ out)
{
    extern __shared__ unsigned short smem[];
    unsigned short* Blds = smem;                    // [j*32+d][s]   : 512 rows
    unsigned short* Plds = Blds + 512*LSTR;         // [i*16+j][k2]  : 256 rows
    unsigned short* Wlds = Plds + 256*LSTR;         // [z][k2]       : 2 x 128 rows
    unsigned short* Alds = Wlds + 2*128*LSTR;       // [c*16+i][s]   : 2 x  64 rows

    const int tid  = threadIdx.x;
    const int wave = tid >> 5, lane = tid & 31;
    const int h = lane >> 4, r = lane & 15;
    const int i0 = blockIdx.x * BI, j0 = blockIdx.y * BJ;

    // Stage B block once: 512 rows x 256B (16 async ops per wave)
    for (int q = tid; q < 512*16; q += 512) {
        int rowq = q >> 4, seg = q & 15;
        copy16_g2l(Blds + rowq*LSTR + seg*8,
                   Bt + ((size_t)j0*C_DIM + rowq)*S_DIM + seg*8);
    }
    // Stage chunk 0 A/W into buffer 0 (6 async ops per wave)
    {
        for (int q = tid; q < 64*16; q += 512) {
            int rowq = q >> 4, seg = q & 15;          // rowq = c*16 + i
            int c = rowq >> 4, i = rowq & 15;
            copy16_g2l(Alds + rowq*LSTR + seg*8,
                       At + ((size_t)(i0+i)*C_DIM + c)*S_DIM + seg*8);
        }
        for (int q = tid; q < 128*16; q += 512) {
            int z = q >> 4, seg = q & 15;
            copy16_g2l(Wlds + z*LSTR + seg*8,
                       Wt + (size_t)z*(C_DIM*C_DIM) + seg*8);
        }
    }

    v8f acc[8] = {};                 // wave owns i_local = wave, all 8 z-tiles

    const int cw    = wave >> 2;     // phase-A: this wave's c within chunk
    const int jbase = (wave & 3)*4;  // phase-A: this wave's 4 j's

    for (int cc = 0; cc < 8; ++cc) {
        __syncthreads();             // prior phase-B done; other buffer free for overwrite

        if (cc < 7) {
            // prefetch chunk cc+1 into the other buffer (6 async ops per wave)
            const int cb2 = (cc+1)*4;
            unsigned short* Al = Alds + ((cc+1)&1)*(64*LSTR);
            unsigned short* Wl = Wlds + ((cc+1)&1)*(128*LSTR);
            for (int q = tid; q < 64*16; q += 512) {
                int rowq = q >> 4, seg = q & 15;
                int c = rowq >> 4, i = rowq & 15;
                copy16_g2l(Al + rowq*LSTR + seg*8,
                           At + ((size_t)(i0+i)*C_DIM + cb2 + c)*S_DIM + seg*8);
            }
            for (int q = tid; q < 128*16; q += 512) {
                int z = q >> 4, seg = q & 15;
                copy16_g2l(Wl + z*LSTR + seg*8,
                           Wt + (size_t)z*(C_DIM*C_DIM) + cb2*C_DIM + seg*8);
            }
            wait_g2l_le6();          // older (chunk cc) copies complete; new 6 in flight
        } else {
            wait_g2l_le0();
        }
        __syncthreads();             // publish chunk cc data

        const unsigned short* Alc = Alds + (cc&1)*(64*LSTR);
        const unsigned short* Wlc = Wlds + (cc&1)*(128*LSTR);

        // -------- Phase A: 8 P-tiles per wave (fixed c=cw; j in jbase..+3; two d halves)
        v16bf aop[4];
#pragma unroll
        for (int k0i = 0; k0i < 4; ++k0i)
            aop[k0i] = load_a_op(Alc + (cw*BI)*LSTR, r, h, k0i*32);
#pragma unroll
        for (int jj = 0; jj < 4; ++jj) {
            const int j = jbase + jj;
#pragma unroll
            for (int dh = 0; dh < 2; ++dh) {
                const int d0 = dh * 16;
                v8f p = {};
#pragma unroll
                for (int k0i = 0; k0i < 4; ++k0i) {
                    v16bf bop = load_b_op(Blds + (j*C_DIM + d0)*LSTR, r, h, k0i*32);
                    p = __builtin_amdgcn_wmma_f32_16x16x32_bf16(
                            false, aop[k0i], false, bop, (short)0, p, false, false);
                }
                // D layout: VGPR v, lane(h,r) -> M=v+8h (=i_local), N=r (=d-d0)
#pragma unroll
                for (int v = 0; v < 8; ++v) {
                    int irow = v + 8*h;
                    Plds[(irow*BJ + j)*LSTR + cw*C_DIM + d0 + r] = f2bfh(p[v]);
                }
            }
        }
        __syncthreads();

        // -------- Phase B: out tile rows = 16 j's of i_local=wave; K = 128 cd's
        v16bf a2[4];
#pragma unroll
        for (int k0i = 0; k0i < 4; ++k0i)
            a2[k0i] = load_a_op(Plds + (wave*BJ)*LSTR, r, h, k0i*32);
#pragma unroll
        for (int zt = 0; zt < 8; ++zt) {
#pragma unroll
            for (int k0i = 0; k0i < 4; ++k0i) {
                v16bf b2 = load_b_op(Wlc + (zt*16)*LSTR, r, h, k0i*32);
                acc[zt] = __builtin_amdgcn_wmma_f32_16x16x32_bf16(
                              false, a2[k0i], false, b2, (short)0, acc[zt], false, false);
            }
        }
    }

    // Epilogue: D layout -> j = v+8h, z = zt*16 + r
    const int ig = i0 + wave;
#pragma unroll
    for (int zt = 0; zt < 8; ++zt) {
        float bias = b_out[zt*16 + r];
#pragma unroll
        for (int v = 0; v < 8; ++v) {
            int jg = j0 + v + 8*h;
            out[((size_t)ig*I_DIM + jg)*CZ + zt*16 + r] = acc[zt][v] + bias;
        }
    }
}

// ---------------------------------------------------------------------------------------
extern "C" void kernel_launch(void* const* d_in, const int* in_sizes, int n_in,
                              void* d_out, int out_size, void* d_ws, size_t ws_size,
                              hipStream_t stream) {
    const float* msa     = (const float*)d_in[0];
    const float* ln_w    = (const float*)d_in[1];
    const float* ln_b    = (const float*)d_in[2];
    const float* w_left  = (const float*)d_in[3];
    const float* b_left  = (const float*)d_in[4];
    const float* w_right = (const float*)d_in[5];
    const float* b_right = (const float*)d_in[6];
    const float* w_out   = (const float*)d_in[7];
    const float* b_out   = (const float*)d_in[8];
    float* out = (float*)d_out;

    unsigned short* At = (unsigned short*)d_ws;                      // 2 MB
    unsigned short* Bt = At + (size_t)S_DIM*I_DIM*C_DIM;             // 2 MB
    unsigned short* Wt = Bt + (size_t)S_DIM*I_DIM*C_DIM;             // 256 KB
    float*         Wlr = (float*)(Wt + (size_t)CZ*C_DIM*C_DIM);     // 64 KB (interleaved wl/wr)

    wlr_pack_kernel<<<dim3((CM*C_DIM)/256), dim3(256), 0, stream>>>(w_left, w_right, Wlr);

    ln_proj_kernel<<<dim3(I_DIM, S_DIM/8), dim3(256), 0, stream>>>(
        msa, ln_w, ln_b, Wlr, b_left, b_right, At, Bt);

    wout_tr_kernel<<<dim3((C_DIM*C_DIM)/32, CZ/32), dim3(256), 0, stream>>>(w_out, Wt);

    const size_t lds_bytes = (size_t)(512 + 256 + 2*128 + 2*64) * LSTR * sizeof(unsigned short); // ~306 KB
    fused_opm_kernel<<<dim3(I_DIM/BI, I_DIM/BJ), dim3(512), lds_bytes, stream>>>(
        At, Bt, Wt, b_out, out);
}
